// ProteinStateReconstructor2D_36103495090325
// MI455X (gfx1250) — matrix operand order repaired and checked by reading, not compiled, not run on hardware
//
#include <hip/hip_runtime.h>
#include <hip/hip_bf16.h>
#include <stdint.h>

// Problem constants (from reference)
#define B_    16
#define N_    16384
#define R_    1024
#define D_    320
#define BINS_ 32
#define H_    512
#define EPS_  1e-5f
#define TILES_ 4            // 128-row tiles per block in the MLP kernel

typedef __attribute__((ext_vector_type(16))) __bf16 v16bf;
typedef __attribute__((ext_vector_type(8)))  float  v8f;

union BFr { v16bf v; uint4 q[2]; };

// ---------------------------------------------------------------------------
// Kernel 1: segment boundaries via binary search (seg_ids are sorted).
// starts[r] = first index n with seg_ids[n] >= r, for r in [0, R]; starts[R]=N.
// ---------------------------------------------------------------------------
__global__ void seg_bounds_kernel(const int* __restrict__ seg_ids,
                                  int* __restrict__ starts) {
    int r = blockIdx.x * blockDim.x + threadIdx.x;
    if (r > R_) return;
    int lo = 0, hi = N_;
    while (lo < hi) {
        int mid = (lo + hi) >> 1;
        if (seg_ids[mid] < r) lo = mid + 1; else hi = mid;
    }
    starts[r] = lo;
}

// ---------------------------------------------------------------------------
// Kernel 2: fold BatchNorm into W2:  W2p[h][j] = s_h * W2[h][j],
// b2p[j] = b2[j] + sum_h (beta_h - mean_h*s_h) * W2[h][j],  s = gamma*rsqrt(var+eps)
// Deterministic fixed-order tree reduction (no float atomics).
// ---------------------------------------------------------------------------
__global__ void fold_bn_kernel(const float* __restrict__ gamma,
                               const float* __restrict__ beta,
                               const float* __restrict__ bn_mean,
                               const float* __restrict__ bn_var,
                               const float* __restrict__ W2,
                               const float* __restrict__ b2,
                               float* __restrict__ W2p,
                               float* __restrict__ b2p) {
    __shared__ float red[3][H_];
    int h = threadIdx.x;  // 512 threads
    float s = gamma[h] * rsqrtf(bn_var[h] + EPS_);
    float t = beta[h] - bn_mean[h] * s;
    float p0 = W2[h * 3 + 0], p1 = W2[h * 3 + 1], p2 = W2[h * 3 + 2];
    W2p[h * 3 + 0] = s * p0;
    W2p[h * 3 + 1] = s * p1;
    W2p[h * 3 + 2] = s * p2;
    red[0][h] = t * p0;
    red[1][h] = t * p1;
    red[2][h] = t * p2;
    __syncthreads();
    for (int off = H_ / 2; off > 0; off >>= 1) {
        if (h < off) {
            red[0][h] += red[0][h + off];
            red[1][h] += red[1][h + off];
            red[2][h] += red[2][h + off];
        }
        __syncthreads();
    }
    if (h < 3) b2p[h] = b2[h] + red[h][0];
}

// ---------------------------------------------------------------------------
// Kernel 3: fused segment-sum + bin pooling (the HBM-roofline leg: 335 MB).
// One block per (segment r, batch b); 320 threads = one channel each, fully
// coalesced. Row loop unrolled x4 + prefetch to keep loads in flight.
// Deterministic: per-channel sums to LDS, fixed-order 10-way bin sum.
// ---------------------------------------------------------------------------
__global__ void pool_kernel(const float* __restrict__ x,
                            const int* __restrict__ starts,
                            float* __restrict__ pooled) {
    __shared__ float chan[D_];
    int r = blockIdx.x, b = blockIdx.y;
    int d = threadIdx.x;  // 0..319
    int s0 = starts[r], s1 = starts[r + 1];
    const float* xp = x + ((size_t)b * N_ + s0) * D_ + d;
    float sum0 = 0.f, sum1 = 0.f, sum2 = 0.f, sum3 = 0.f;
    int n = s0;
    for (; n + 3 < s1; n += 4) {
        __builtin_prefetch(xp + 16 * D_, 0, 0);   // ~16 rows (20 KB) ahead
        float a0 = xp[0 * D_];
        float a1 = xp[1 * D_];
        float a2 = xp[2 * D_];
        float a3 = xp[3 * D_];
        sum0 += a0; sum1 += a1; sum2 += a2; sum3 += a3;
        xp += 4 * D_;
    }
    for (; n < s1; ++n) { sum0 += *xp; xp += D_; }
    chan[d] = (sum0 + sum1) + (sum2 + sum3);
    __syncthreads();
    if (d < BINS_) {
        const float* cp = chan + d * 10;
        float s = (((cp[0] + cp[1]) + (cp[2] + cp[3])) +
                   ((cp[4] + cp[5]) + (cp[6] + cp[7]))) + (cp[8] + cp[9]);
        float cnt = (float)(s1 - s0);
        pooled[((size_t)b * R_ + r) * BINS_ + d] = s / (10.f * fmaxf(cnt, 1.f));
    }
}

// ---------------------------------------------------------------------------
// Kernel 4: WMMA MLP. 256 threads = 8 waves; each block processes TILES_ x 128
// rows (W1 staged to LDS once per 512 rows). Per wave: 16 rows.
// z = feat @ W1 + b1 via v_wmma_f32_16x16x32_bf16 (K=32 = pooled bins; the 3
// cond columns folded into the C operand). ReLU, then the BN-folded 512->3
// GEMM done per-lane on the WMMA D tile + shfl-xor reduction. h never hits
// memory.
// ---------------------------------------------------------------------------
__global__ void __launch_bounds__(256)
mlp_kernel(const float* __restrict__ pooled,
           const float* __restrict__ cond,
           const int* __restrict__ seg_ids,
           const float* __restrict__ W1,
           const float* __restrict__ b1,
           const float* __restrict__ W2p,
           const float* __restrict__ b2p,
           float* __restrict__ out) {
    // W1 (k<32) transposed to bf16, column stride 5 uint4 = 80 B.
    // Lane fragment loads are 16B-aligned uint4s; stride 80B = 20 banks ->
    // lane start banks {0,4,...,60}: conflict-free ds_load_b128.
    __shared__ uint4 w1q[H_ * 5];                     // 40 KB
    __shared__ float w1clds[3 * H_];                  // 6 KB   W1 rows 32..34 (cond)
    __shared__ float w2plds[H_ * 3];                  // 6 KB   folded W2
    __shared__ float b1lds[H_];                       // 2 KB
    __shared__ float condlds[128 * 4];                // 2 KB

    int tid = threadIdx.x;
    long rowbase0 = (long)blockIdx.x * (128 * TILES_);

    // Stage W1 (transpose k-major -> column-major bf16): coalesced global reads.
    __bf16* w1b = (__bf16*)w1q;
    for (int k = 0; k < BINS_; ++k)
        for (int h = tid; h < H_; h += 256)
            w1b[h * 40 + k] = (__bf16)W1[k * H_ + h];
    for (int t = tid; t < 3 * H_; t += 256)
        w1clds[t] = W1[(BINS_ + t / H_) * H_ + (t % H_)];
    for (int t = tid; t < H_ * 3; t += 256) w2plds[t] = W2p[t];
    for (int t = tid; t < H_; t += 256)     b1lds[t]  = b1[t];

    int wave = tid >> 5, lane = tid & 31;
    int half = lane >> 4, l16 = lane & 15;
    float bp0 = b2p[0], bp1 = b2p[1], bp2 = b2p[2];

    for (int tile = 0; tile < TILES_; ++tile) {
        long rowbase = rowbase0 + tile * 128;
        __syncthreads();   // staged data ready / condlds safe to overwrite
        for (int t = tid; t < 128; t += 256) {
            int n = (int)((rowbase + t) & (N_ - 1));
            condlds[t * 4 + 0] = cond[n * 3 + 0];
            condlds[t * 4 + 1] = cond[n * 3 + 1];
            condlds[t * 4 + 2] = cond[n * 3 + 2];
            condlds[t * 4 + 3] = 0.f;
        }
        __syncthreads();

        long wave_base = rowbase + wave * 16;

        // Build A fragment (16x32 bf16, CDNA5 layout: lane m=l16 holds row m;
        // lanes 0-15 hold K={0..7,16..23}, lanes 16-31 K={8..15,24..31}).
        long g = wave_base + l16;
        int b = (int)(g >> 14);          // / N_
        int n = (int)(g & (N_ - 1));
        int sgm = seg_ids[n];
        const float* prow = pooled + ((size_t)b * R_ + sgm) * BINS_;
        int k0 = half * 8;
        BFr A;
#pragma unroll
        for (int e = 0; e < 8; ++e) {
            A.v[e]     = (__bf16)prow[k0 + e];
            A.v[8 + e] = (__bf16)prow[16 + k0 + e];
        }

        // cond values for this lane's 8 output rows (D tile: VGPR v -> m=v+8*half)
        float c0[8], c1[8], c2[8];
#pragma unroll
        for (int v = 0; v < 8; ++v) {
            int rl = wave * 16 + v + half * 8;
            c0[v] = condlds[rl * 4 + 0];
            c1[v] = condlds[rl * 4 + 1];
            c2[v] = condlds[rl * 4 + 2];
        }

        float oacc[8][3];
#pragma unroll
        for (int v = 0; v < 8; ++v) { oacc[v][0] = 0.f; oacc[v][1] = 0.f; oacc[v][2] = 0.f; }

#pragma unroll 2
        for (int c = 0; c < H_ / 16; ++c) {      // 32 N-chunks of 16 columns
            int ncol = c * 16 + l16;
            // B fragment (32x16 bf16): lane col n=ncol, K = {half*8 + 0..7, 16 + half*8 + 0..7}
            BFr Bf;
            int qi = ncol * 5 + half;
            Bf.q[0] = w1q[qi];
            Bf.q[1] = w1q[qi + 2];

            float bb  = b1lds[ncol];
            float w10 = w1clds[ncol], w11 = w1clds[H_ + ncol], w12 = w1clds[2 * H_ + ncol];
            v8f cini;
#pragma unroll
            for (int v = 0; v < 8; ++v)
                cini[v] = bb + c0[v] * w10 + c1[v] * w11 + c2[v] * w12;

            v8f acc = __builtin_amdgcn_wmma_f32_16x16x32_bf16(
                false, A.v, false, Bf.v, (short)0, cini, false, false);

            float w20 = w2plds[ncol * 3 + 0];
            float w21 = w2plds[ncol * 3 + 1];
            float w22 = w2plds[ncol * 3 + 2];
#pragma unroll
            for (int v = 0; v < 8; ++v) {
                float rz = acc[v] > 0.f ? acc[v] : 0.f;   // ReLU on z tile
                oacc[v][0] += rz * w20;
                oacc[v][1] += rz * w21;
                oacc[v][2] += rz * w22;
            }
        }

        // Reduce over the 16 columns held by the 16 lanes of this half.
#pragma unroll
        for (int mask = 1; mask < 16; mask <<= 1) {
#pragma unroll
            for (int v = 0; v < 8; ++v) {
                oacc[v][0] += __shfl_xor(oacc[v][0], mask, 32);
                oacc[v][1] += __shfl_xor(oacc[v][1], mask, 32);
                oacc[v][2] += __shfl_xor(oacc[v][2], mask, 32);
            }
        }

        if (l16 < 8) {
            float o0 = 0.f, o1 = 0.f, o2 = 0.f;
#pragma unroll
            for (int v = 0; v < 8; ++v)
                if (v == l16) { o0 = oacc[v][0]; o1 = oacc[v][1]; o2 = oacc[v][2]; }
            long row = wave_base + l16 + half * 8;
            out[row * 3 + 0] = o0 + bp0;
            out[row * 3 + 1] = o1 + bp1;
            out[row * 3 + 2] = o2 + bp2;
        }
    }
}

// ---------------------------------------------------------------------------
extern "C" void kernel_launch(void* const* d_in, const int* in_sizes, int n_in,
                              void* d_out, int out_size, void* d_ws, size_t ws_size,
                              hipStream_t stream) {
    const float* x       = (const float*)d_in[0];
    const float* cond    = (const float*)d_in[1];
    const int*   seg_ids = (const int*)  d_in[2];
    const float* W1      = (const float*)d_in[3];
    const float* b1      = (const float*)d_in[4];
    const float* gamma   = (const float*)d_in[5];
    const float* beta    = (const float*)d_in[6];
    const float* bn_mean = (const float*)d_in[7];
    const float* bn_var  = (const float*)d_in[8];
    const float* W2      = (const float*)d_in[9];
    const float* b2      = (const float*)d_in[10];
    float* out = (float*)d_out;

    char* ws = (char*)d_ws;
    int*   starts = (int*)  (ws + 0);        // (R+1) ints
    float* W2p    = (float*)(ws + 8192);     // H*3 floats
    float* b2p    = (float*)(ws + 16384);    // 3 floats
    float* pooled = (float*)(ws + 32768);    // B*R*BINS floats (2 MB)

    seg_bounds_kernel<<<(R_ + 1 + 255) / 256, 256, 0, stream>>>(seg_ids, starts);
    fold_bn_kernel<<<1, H_, 0, stream>>>(gamma, beta, bn_mean, bn_var, W2, b2, W2p, b2p);
    pool_kernel<<<dim3(R_, B_), D_, 0, stream>>>(x, starts, pooled);
    mlp_kernel<<<(B_ * N_) / (128 * TILES_), 256, 0, stream>>>(
        pooled, cond, seg_ids, W1, b1, W2p, b2p, out);
}
